// TransformerBlock_9457517986497
// MI455X (gfx1250) — matrix operand order repaired and checked by reading, not compiled, not run on hardware
//
#include <hip/hip_runtime.h>
#include <hip/hip_bf16.h>

// ---------------------------------------------------------------------------
// Transformer block on gfx1250 (MI455X): bf16 WMMA + double-buffered async
// global->LDS staging (GLOBAL_LOAD_ASYNC_TO_LDS_B128 / ASYNCcnt).
// All GEMMs / attention einsums on v_wmma_f32_16x16x32_bf16; fp32 spine.
// ---------------------------------------------------------------------------

typedef __bf16 bf16;
typedef __attribute__((ext_vector_type(8)))  __bf16 bf16x8;
typedef __attribute__((ext_vector_type(16))) __bf16 v16bf;
typedef __attribute__((ext_vector_type(8)))  float  v8f;

#define BB   2
#define SS   2048
#define DD   1024
#define HH   16
#define HDIM 64
#define DFFN 4096
#define MTOT (BB * SS)   // 4096 rows

static __device__ inline v8f vzero8() {
  v8f r;
#pragma unroll
  for (int i = 0; i < 8; ++i) r[i] = 0.0f;
  return r;
}

static __device__ inline v16bf cat16(bf16x8 lo, bf16x8 hi) {
  v16bf r;
#pragma unroll
  for (int i = 0; i < 8; ++i) { r[i] = lo[i]; r[i + 8] = hi[i]; }
  return r;
}

static __device__ inline v8f wmma_bf16(v16bf a, v16bf b, v8f c) {
  // 8 args: (neg_a, A, neg_b, B, c_mod, C, reuse_a, reuse_b)
  return __builtin_amdgcn_wmma_f32_16x16x32_bf16(false, a, false, b,
                                                 (short)0, c, false, false);
}

// Async 16-byte global -> LDS copy (GLOBAL_LOAD_ASYNC_TO_LDS_B128, ASYNCcnt).
// Low 32 bits of a generic pointer into LDS are the LDS byte address.
static __device__ inline void cp_async16(void* lds_dst, const void* gsrc) {
  unsigned l = (unsigned)(unsigned long long)lds_dst;
  unsigned long long g = (unsigned long long)gsrc;
  asm volatile("global_load_async_to_lds_b128 %0, %1, off"
               :: "v"(l), "v"(g) : "memory");
}
static __device__ inline void async_wait0() {
  asm volatile("s_wait_asynccnt 0x0" ::: "memory");
}

// ---------------------------------------------------------------------------
// fp32 -> bf16 transposing weight conversion: out[N][K] = (bf16) in[K][N]
// ---------------------------------------------------------------------------
__global__ __launch_bounds__(256) void cvt_t_kernel(const float* __restrict__ in,
                                                    bf16* __restrict__ out,
                                                    int K, int N) {
  __shared__ float tile[16][17];
  const int n0 = blockIdx.x * 16;
  const int k0 = blockIdx.y * 16;
  const int tx = threadIdx.x & 15;
  const int ty = threadIdx.x >> 4;
  tile[ty][tx] = in[(size_t)(k0 + ty) * N + n0 + tx];
  __syncthreads();
  out[(size_t)(n0 + ty) * K + k0 + tx] = (bf16)tile[tx][ty];
}

// ---------------------------------------------------------------------------
// LayerNorm: one block per row (D = 1024), fp32 in, bf16 out
// ---------------------------------------------------------------------------
__global__ __launch_bounds__(256) void ln_kernel(const float* __restrict__ x,
                                                 const float* __restrict__ g,
                                                 const float* __restrict__ beta,
                                                 bf16* __restrict__ out) {
  size_t row = blockIdx.x;
  int tid = threadIdx.x;
  const float* xr = x + row * DD;
  float lv[4];
  float s = 0.0f, s2 = 0.0f;
#pragma unroll
  for (int i = 0; i < 4; ++i) {
    float v = xr[tid + i * 256];
    lv[i] = v;
    s += v;
    s2 += v * v;
  }
  __shared__ float rs[256];
  __shared__ float rq[256];
  rs[tid] = s; rq[tid] = s2;
  __syncthreads();
  for (int off = 128; off > 0; off >>= 1) {
    if (tid < off) { rs[tid] += rs[tid + off]; rq[tid] += rq[tid + off]; }
    __syncthreads();
  }
  float mu   = rs[0] * (1.0f / DD);
  float var  = rq[0] * (1.0f / DD) - mu * mu;
  float rstd = rsqrtf(var + 1e-5f);
#pragma unroll
  for (int i = 0; i < 4; ++i) {
    int c = tid + i * 256;
    out[row * DD + c] = (bf16)((lv[i] - mu) * rstd * g[c] + beta[c]);
  }
}

// ---------------------------------------------------------------------------
// Tiled WMMA GEMM: C[MxN] = act( A[MxK](bf16) * Bt[NxK](bf16)^T + bias + res )
// Bt is the pre-transposed weight. block = 128 threads (4 waves).
// Block tile 128x64; wave tile 32x64 (2 A-frags x 4 B-frags = 8 wmma/K-step).
// Double-buffered async staging: next K-tile DMA overlaps current WMMAs.
// OUT_MODE: 0 = fp32 row-major, 1 = bf16 row-major, 2 = bf16 [B,H,hd,S] (V^T)
// ---------------------------------------------------------------------------
template <bool GELU, int OUT_MODE>
__global__ __launch_bounds__(128) void gemm_kernel(const bf16* __restrict__ A,
                                                   const bf16* __restrict__ Bt,
                                                   const float* __restrict__ bias,
                                                   const float* __restrict__ res,
                                                   void* __restrict__ outv,
                                                   int M, int N, int K) {
  __shared__ __align__(16) bf16 As[2][128][40];   // [m][k] + pad
  __shared__ __align__(16) bf16 Bs[2][64][40];    // [n][k] + pad

  const int n0 = blockIdx.x * 64;
  const int m0 = blockIdx.y * 128;
  const int tid  = threadIdx.x;
  const int w    = tid >> 5;
  const int lane = tid & 31;
  const int lo16 = lane & 15;
  const bool lh  = lane < 16;

  v8f acc[2][4];
#pragma unroll
  for (int s = 0; s < 2; ++s)
#pragma unroll
    for (int t = 0; t < 4; ++t) acc[s][t] = vzero8();

  const int brow = tid >> 1;          // 0..63
  const int bcb  = (tid & 1) * 16;    // 0/16

  // async-stage one 128x32 A tile + 64x32 B tile into buffer `buf`
  auto issue = [&](int k0, int buf) {
    const bf16* ap = A + (size_t)(m0 + tid) * K + k0;       // row = tid (0..127)
#pragma unroll
    for (int c = 0; c < 4; ++c)
      cp_async16(&As[buf][tid][c * 8], ap + c * 8);
    const bf16* bp = Bt + (size_t)(n0 + brow) * K + k0 + bcb;
    cp_async16(&Bs[buf][brow][bcb],     bp);
    cp_async16(&Bs[buf][brow][bcb + 8], bp + 8);
  };

  issue(0, 0);
  int cur = 0;
  const int ab = lh ? 0 : 8;          // A K-pattern base
  const int kb = lh ? 0 : 16;         // unused; B frags also use ab

  for (int k0 = 0; k0 < K; k0 += 32) {
    async_wait0();
    __syncthreads();
    if (k0 + 32 < K) issue(k0 + 32, cur ^ 1);   // overlap next DMA with wmma

    v16bf af0 = cat16(*(const bf16x8*)&As[cur][w * 32 + lo16][ab],
                      *(const bf16x8*)&As[cur][w * 32 + lo16][ab + 16]);
    v16bf af1 = cat16(*(const bf16x8*)&As[cur][w * 32 + 16 + lo16][ab],
                      *(const bf16x8*)&As[cur][w * 32 + 16 + lo16][ab + 16]);
#pragma unroll
    for (int t = 0; t < 4; ++t) {
      const bf16* bq = &Bs[cur][t * 16 + lo16][ab];
      v16bf bf = cat16(*(const bf16x8*)bq, *(const bf16x8*)(bq + 16));
      acc[0][t] = wmma_bf16(af0, bf, acc[0][t]);
      acc[1][t] = wmma_bf16(af1, bf, acc[1][t]);
    }
    __syncthreads();
    cur ^= 1;
  }

  // ---- epilogue: bias / residual / exact GELU / store ----
#pragma unroll
  for (int s = 0; s < 2; ++s) {
#pragma unroll
    for (int t = 0; t < 4; ++t) {
      const int col = n0 + t * 16 + lo16;
      const float bv = bias ? bias[col] : 0.0f;
#pragma unroll
      for (int r = 0; r < 8; ++r) {
        const int row = m0 + w * 32 + s * 16 + r + (lh ? 0 : 8);
        float v = acc[s][t][r] + bv;
        if (res) v += res[(size_t)row * N + col];
        if (GELU) v = 0.5f * v * (1.0f + erff(v * 0.70710678118654752f));
        if (OUT_MODE == 0) {
          ((float*)outv)[(size_t)row * N + col] = v;
        } else if (OUT_MODE == 1) {
          ((bf16*)outv)[(size_t)row * N + col] = (bf16)v;
        } else {
          // V^T layout [B, H, hd, S] for vectorized attention staging
          const int b = row >> 11;          // row / S
          const int ss = row & (SS - 1);
          const int h = col >> 6;           // col / HDIM
          const int d = col & (HDIM - 1);
          ((bf16*)outv)[(((size_t)b * HH + h) * HDIM + d) * SS + ss] = (bf16)v;
        }
      }
    }
  }
}

// ---------------------------------------------------------------------------
// Causal flash attention, bf16 WMMA, fp32 softmax state.
// grid = (S/64, H, B); block = 128 threads = 4 waves; wave = 16 query rows.
// K staged [key][hd], V staged [hd][key] from V^T — double-buffered async.
// ---------------------------------------------------------------------------
__global__ __launch_bounds__(128) void attn_kernel(const bf16* __restrict__ qg,
                                                   const bf16* __restrict__ kg,
                                                   const bf16* __restrict__ vtg,
                                                   bf16* __restrict__ ctxg) {
  __shared__ __align__(16) bf16 Ks[2][32][72];     // [key][hd] + pad
  __shared__ __align__(16) bf16 Vs[2][64][40];     // [hd][key] + pad
  __shared__ __align__(16) bf16 Ps[4][16][40];     // per-wave probs [row][key]

  const int q0 = blockIdx.x * 64;
  const int h  = blockIdx.y;
  const int b  = blockIdx.z;
  const int tid  = threadIdx.x;
  const int w    = tid >> 5;
  const int lane = tid & 31;
  const int lo16 = lane & 15;
  const bool lh  = lane < 16;
  const int ab   = lh ? 0 : 8;                     // A-frag key-pattern base
  const int kb   = lh ? 0 : 16;                    // unused; B frags also use ab

  const size_t base  = ((size_t)b * SS) * DD + (size_t)h * HDIM;   // q/k/ctx
  const size_t vbase = ((size_t)b * HH + h) * (size_t)(HDIM * SS); // V^T

  // ---- persistent Q fragments (hd split 0-31 / 32-63) ----
  const int qrow = q0 + w * 16 + lo16;
  const bf16* qrp = qg + base + (size_t)qrow * DD;
  v16bf qfrag[2];
#pragma unroll
  for (int c = 0; c < 2; ++c)
    qfrag[c] = cat16(*(const bf16x8*)(qrp + c * 32 + ab),
                     *(const bf16x8*)(qrp + c * 32 + ab + 16));

  v8f acc[4];
#pragma unroll
  for (int t = 0; t < 4; ++t) acc[t] = vzero8();
  float m[8], l[8];
#pragma unroll
  for (int r = 0; r < 8; ++r) { m[r] = -3.0e38f; l[r] = 0.0f; }

  const int wrmax  = q0 + w * 16 + 15;             // this wave's last query row
  const int key_ld = tid >> 2;                     // 0..31  (K staging)
  const int cb_ld  = (tid & 3) * 16;               // 0/16/32/48
  const int d_ld   = tid >> 1;                     // 0..63  (V staging)
  const int kc_ld  = (tid & 1) * 16;               // 0/16

  auto issue_kv = [&](int j0, int buf) {
    const bf16* kp = kg + base + (size_t)(j0 + key_ld) * DD + cb_ld;
    cp_async16(&Ks[buf][key_ld][cb_ld],     kp);
    cp_async16(&Ks[buf][key_ld][cb_ld + 8], kp + 8);
    const bf16* vp = vtg + vbase + (size_t)d_ld * SS + j0 + kc_ld;
    cp_async16(&Vs[buf][d_ld][kc_ld],     vp);
    cp_async16(&Vs[buf][d_ld][kc_ld + 8], vp + 8);
  };

  issue_kv(0, 0);
  int cur = 0;

  for (int j0 = 0; j0 <= q0 + 63; j0 += 32) {
    async_wait0();
    __syncthreads();
    if (j0 + 32 <= q0 + 63) issue_kv(j0 + 32, cur ^ 1);  // overlap next DMA

    if (j0 <= wrmax) {                             // causally relevant for wave
      // ---- scores: two 16-key tiles, each QK^T over hd=64 (2 wmma) ----
      v8f st[2];
#pragma unroll
      for (int hj = 0; hj < 2; ++hj) {
        v8f s = vzero8();
#pragma unroll
        for (int c = 0; c < 2; ++c) {
          const bf16* bq = &Ks[cur][hj * 16 + lo16][c * 32 + ab];
          v16bf bf = cat16(*(const bf16x8*)bq, *(const bf16x8*)(bq + 16));
          s = wmma_bf16(qfrag[c], bf, s);
        }
        st[hj] = s;
      }
      // ---- online softmax (row lives in elem r across a 16-lane half) ----
      float corr[8];
#pragma unroll
      for (int r = 0; r < 8; ++r) {
        const int row = q0 + w * 16 + r + (lh ? 0 : 8);
        float x0 = st[0][r] * 0.125f;              // 1/sqrt(64)
        float x1 = st[1][r] * 0.125f;
        if (j0 + lo16 > row)      x0 = -3.0e38f;
        if (j0 + 16 + lo16 > row) x1 = -3.0e38f;
        float tmax = fmaxf(x0, x1);
#pragma unroll
        for (int d = 1; d < 16; d <<= 1) tmax = fmaxf(tmax, __shfl_xor(tmax, d, 32));
        const float nm = fmaxf(m[r], tmax);
        corr[r] = __expf(m[r] - nm);
        const float p0 = __expf(x0 - nm);
        const float p1 = __expf(x1 - nm);
        float psum = p0 + p1;
#pragma unroll
        for (int d = 1; d < 16; d <<= 1) psum += __shfl_xor(psum, d, 32);
        l[r] = l[r] * corr[r] + psum;
        m[r] = nm;
        const int rr = r + (lh ? 0 : 8);
        Ps[w][rr][lo16]      = (bf16)p0;
        Ps[w][rr][16 + lo16] = (bf16)p1;
#pragma unroll
        for (int t = 0; t < 4; ++t) acc[t][r] *= corr[r];
      }
      // ---- ctx += P(16x32) * V(32x64): reload P in A-frag layout ----
      const bf16* pp = &Ps[w][lo16][0];
      v16bf pfrag = cat16(*(const bf16x8*)(pp + ab),
                          *(const bf16x8*)(pp + ab + 16));
#pragma unroll
      for (int t = 0; t < 4; ++t) {
        const bf16* vq = &Vs[cur][t * 16 + lo16][ab];
        v16bf vf = cat16(*(const bf16x8*)vq, *(const bf16x8*)(vq + 16));
        acc[t] = wmma_bf16(pfrag, vf, acc[t]);
      }
    }
    __syncthreads();
    cur ^= 1;
  }

  // ---- normalize and write ctx (bf16, [B,S,D] head-interleaved) ----
#pragma unroll
  for (int r = 0; r < 8; ++r) {
    const int srow = q0 + w * 16 + r + (lh ? 0 : 8);
    const float inv = 1.0f / l[r];
#pragma unroll
    for (int t = 0; t < 4; ++t)
      ctxg[base + (size_t)srow * DD + t * 16 + lo16] = (bf16)(acc[t][r] * inv);
  }
}

// ---------------------------------------------------------------------------
// Launch
// ---------------------------------------------------------------------------
extern "C" void kernel_launch(void* const* d_in, const int* in_sizes, int n_in,
                              void* d_out, int out_size, void* d_ws, size_t ws_size,
                              hipStream_t stream) {
  const float* x     = (const float*)d_in[0];
  const float* ln1_g = (const float*)d_in[1];
  const float* ln1_b = (const float*)d_in[2];
  const float* Wq    = (const float*)d_in[3];
  const float* Wk    = (const float*)d_in[4];
  const float* Wv    = (const float*)d_in[5];
  const float* Wo    = (const float*)d_in[6];
  const float* bo    = (const float*)d_in[7];
  const float* ln2_g = (const float*)d_in[8];
  const float* ln2_b = (const float*)d_in[9];
  const float* W1    = (const float*)d_in[10];
  const float* b1    = (const float*)d_in[11];
  const float* W2    = (const float*)d_in[12];
  const float* b2    = (const float*)d_in[13];

  char* w = (char*)d_ws;
  const size_t MB = 1u << 20;
  bf16*  h1   = (bf16*)(w + 0 * MB);     // 8 MiB  : ln1 out
  bf16*  qb   = (bf16*)(w + 8 * MB);     // 8 MiB
  bf16*  kbuf = (bf16*)(w + 16 * MB);    // 8 MiB
  bf16*  vtb  = (bf16*)(w + 24 * MB);    // 8 MiB  : V^T [B,H,hd,S]
  bf16*  ctx  = (bf16*)(w + 32 * MB);    // 8 MiB
  float* x1   = (float*)(w + 40 * MB);   // 16 MiB : attn residual out
  bf16*  h2   = (bf16*)(w + 56 * MB);    // 8 MiB  : ln2 out
  bf16*  ffn  = (bf16*)(w + 64 * MB);    // 32 MiB : GELU(h2@W1+b1)
  bf16*  Wqt  = (bf16*)(w + 96 * MB);    // 2 MiB each, transposed [N][K]
  bf16*  Wkt  = (bf16*)(w + 98 * MB);
  bf16*  Wvt  = (bf16*)(w + 100 * MB);
  bf16*  Wot  = (bf16*)(w + 102 * MB);
  bf16*  W1t  = (bf16*)(w + 104 * MB);   // 8 MiB
  bf16*  W2t  = (bf16*)(w + 112 * MB);   // 8 MiB

  // transpose + convert weights: out[N][K] = bf16(in[K][N])
  cvt_t_kernel<<<dim3(DD / 16, DD / 16),   256, 0, stream>>>(Wq, Wqt, DD, DD);
  cvt_t_kernel<<<dim3(DD / 16, DD / 16),   256, 0, stream>>>(Wk, Wkt, DD, DD);
  cvt_t_kernel<<<dim3(DD / 16, DD / 16),   256, 0, stream>>>(Wv, Wvt, DD, DD);
  cvt_t_kernel<<<dim3(DD / 16, DD / 16),   256, 0, stream>>>(Wo, Wot, DD, DD);
  cvt_t_kernel<<<dim3(DFFN / 16, DD / 16), 256, 0, stream>>>(W1, W1t, DD, DFFN);
  cvt_t_kernel<<<dim3(DD / 16, DFFN / 16), 256, 0, stream>>>(W2, W2t, DFFN, DD);

  // pre-LN 1
  ln_kernel<<<MTOT, 256, 0, stream>>>(x, ln1_g, ln1_b, h1);

  // QKV projections (no bias); V written transposed for attention staging
  dim3 gP(DD / 64, MTOT / 128);
  gemm_kernel<false, 1><<<gP, 128, 0, stream>>>(h1, Wqt, nullptr, nullptr, qb,   MTOT, DD, DD);
  gemm_kernel<false, 1><<<gP, 128, 0, stream>>>(h1, Wkt, nullptr, nullptr, kbuf, MTOT, DD, DD);
  gemm_kernel<false, 2><<<gP, 128, 0, stream>>>(h1, Wvt, nullptr, nullptr, vtb,  MTOT, DD, DD);

  // causal attention
  attn_kernel<<<dim3(SS / 64, HH, BB), 128, 0, stream>>>(qb, kbuf, vtb, ctx);

  // out projection + bias + residual (fp32)
  gemm_kernel<false, 0><<<gP, 128, 0, stream>>>(ctx, Wot, bo, x, x1, MTOT, DD, DD);

  // pre-LN 2
  ln_kernel<<<MTOT, 256, 0, stream>>>(x1, ln2_g, ln2_b, h2);

  // FFN1 + exact GELU
  dim3 gF1(DFFN / 64, MTOT / 128);
  gemm_kernel<true, 1><<<gF1, 128, 0, stream>>>(h2, W1t, b1, nullptr, ffn, MTOT, DFFN, DD);

  // FFN2 + bias + residual -> fp32 output
  gemm_kernel<false, 0><<<gP, 128, 0, stream>>>(ffn, W2t, b2, x1, (float*)d_out, MTOT, DD, DFFN);
}
